// RNN_10290741641474
// MI455X (gfx1250) — matrix operand order repaired and checked by reading, not compiled, not run on hardware
//
#include <hip/hip_runtime.h>
#include <hip/hip_bf16.h>

// ---------------------------------------------------------------------------
// Persistent 3-layer LSTM for MI455X (gfx1250).
//  - f16 WMMA (v_wmma_f32_16x16x32_f16), f32 accumulate, cell state in VGPRs
//  - all weights repacked to WMMA B-layout and pinned in LDS (320KB/WGP exact)
//  - 32 workgroups x 256 threads, grid barrier per layer per timestep
// ---------------------------------------------------------------------------

#define T_STEPS 512
#define BATCH   64
#define NIN     256
#define NHID    512
#define NOUT    256
#define NWG     32
#define WGSIZE  256

typedef _Float16 half_t;
typedef __attribute__((ext_vector_type(16))) _Float16 v16h;
typedef __attribute__((ext_vector_type(8)))  _Float16 v8h;
typedef __attribute__((ext_vector_type(8)))  float    v8f;
typedef __attribute__((ext_vector_type(4)))  float    v4f;

// K-slice counts (K/32) per layer (inputs are [x|h] concatenated along K)
#define KS1 24   // layer1: 256 + 512
#define KS2 32   // layer2: 512 + 512
#define KS3 24   // layer3: 512 + 256

// flat output offsets (reference returns: out, (h1,c1),(h2,c2),(h3,c3))
#define OFF_H1 ((size_t)T_STEPS * BATCH * NOUT)
#define OFF_C1 (OFF_H1 + (size_t)BATCH * NHID)
#define OFF_H2 (OFF_C1 + (size_t)BATCH * NHID)
#define OFF_C2 (OFF_H2 + (size_t)BATCH * NHID)
#define OFF_H3 (OFF_C2 + (size_t)BATCH * NHID)
#define OFF_C3 (OFF_H3 + (size_t)BATCH * NOUT)

// LDS bank-conflict swizzle for B fragments: rotate lane groups of 8
__device__ __host__ __forceinline__ int swz_lane(int l) {
    return (l & 24) | ((l + (l >> 3)) & 7);
}

__device__ __forceinline__ v16h combine16(v8h lo, v8h hi) {
    return __builtin_shufflevector(lo, hi, 0,1,2,3,4,5,6,7,8,9,10,11,12,13,14,15);
}

// A-fragment (16x32 f16, M x K) from row-major f16 [rows x ld].
// ISA layout: lanes 0-15 hold row (lane&15), K {0..7,16..23}; lanes 16-31 same
// row, K {8..15,24..31}  ->  two 16B loads per lane.
__device__ __forceinline__ v16h load_a_f16(const half_t* base, int ld, int row,
                                           int k0, int lane) {
    const half_t* p = base + (size_t)row * ld + k0 + ((lane & 16) ? 8 : 0);
    v8h lo = *(const v8h*)(p);
    v8h hi = *(const v8h*)(p + 16);
    return combine16(lo, hi);
}

// Same A-fragment but sourced from f32 row-major data (seq), convert to f16.
__device__ __forceinline__ v16h load_a_f32(const float* base, int ld, int row,
                                           int k0, int lane) {
    const float* p = base + (size_t)row * ld + k0 + ((lane & 16) ? 8 : 0);
    v4f a0 = *(const v4f*)(p);
    v4f a1 = *(const v4f*)(p + 4);
    v4f b0 = *(const v4f*)(p + 16);
    v4f b1 = *(const v4f*)(p + 20);
    v16h r;
#pragma unroll
    for (int i = 0; i < 4; ++i) {
        r[i]      = (half_t)a0[i];
        r[4 + i]  = (half_t)a1[i];
        r[8 + i]  = (half_t)b0[i];
        r[12 + i] = (half_t)b1[i];
    }
    return r;
}

// B-fragment (32x16 f16, K x N) from a 1KB pre-swizzled LDS block.
__device__ __forceinline__ v16h load_b_lds(const half_t* blk, int lane) {
    const half_t* p = blk + swz_lane(lane) * 16;
    v8h lo = *(const v8h*)(p);      // ds_load_b128
    v8h hi = *(const v8h*)(p + 8);  // ds_load_b128
    return combine16(lo, hi);
}

__device__ __forceinline__ v8f wmma_step(v16h a, v16h b, v8f c) {
    return __builtin_amdgcn_wmma_f32_16x16x32_f16(false, a, false, b,
                                                  (short)0, c, false, false);
}

__device__ __forceinline__ v8f acc_zero() {
    v8f z;
#pragma unroll
    for (int i = 0; i < 8; ++i) z[i] = 0.0f;
    return z;
}

__device__ __forceinline__ float sigf(float x) {
    return 1.0f / (1.0f + __expf(-x));
}
__device__ __forceinline__ float tanh_fast(float x) {
    float e = __expf(-2.0f * x);
    return (1.0f - e) / (1.0f + e);
}

// Device-scope sense-free barrier: monotonically increasing counter.
__device__ __forceinline__ void grid_sync(unsigned* bar, unsigned* target) {
    __syncthreads();
    __threadfence();                      // release: flush WGP caches
    *target += NWG;
    if (threadIdx.x == 0) {
        __hip_atomic_fetch_add(bar, 1u, __ATOMIC_ACQ_REL, __HIP_MEMORY_SCOPE_AGENT);
        while (__hip_atomic_load(bar, __ATOMIC_ACQUIRE, __HIP_MEMORY_SCOPE_AGENT) < *target)
            __builtin_amdgcn_s_sleep(2);
    }
    __syncthreads();
    __threadfence();                      // acquire: invalidate stale lines
}

// ---------------------------------------------------------------------------
// Weight repack: f32 [4H x K_ih]/[4H x K_hh] -> f16 WMMA-B tiles, LDS-swizzled.
// Layout: [n_tile][ks][512 halves]; within a block, swizzled lane slot sl
// stores column n = nt*16+(l&15), K = ks*32 + ((l&16)?8:0) + (j&7) + ((j&8)?16:0).
// ---------------------------------------------------------------------------
__global__ void repack_weights(const float* __restrict__ Wih,
                               const float* __restrict__ Whh,
                               int Kih, int Khh, int nTiles,
                               half_t* __restrict__ dst) {
    int KS = (Kih + Khh) / 32;
    size_t total = (size_t)nTiles * KS * 512;
    size_t idx = (size_t)blockIdx.x * blockDim.x + threadIdx.x;
    if (idx >= total) return;
    int    r   = (int)(idx & 511);
    size_t blk = idx >> 9;
    int sl = r >> 4;
    int j  = r & 15;
    int l  = (sl & 24) | ((sl - (sl >> 3)) & 7);   // inverse of swz_lane
    int ks = (int)(blk % KS);
    int nt = (int)(blk / KS);
    int k  = ks * 32 + ((l & 16) ? 8 : 0) + (j & 7) + ((j & 8) ? 16 : 0);
    int n  = nt * 16 + (l & 15);
    float v = (k < Kih) ? Wih[(size_t)n * Kih + k]
                        : Whh[(size_t)n * Khh + (k - Kih)];
    dst[idx] = (half_t)v;
}

__global__ void bias_sum(const float* __restrict__ a, const float* __restrict__ b,
                         float* __restrict__ dst, int n) {
    int i = blockIdx.x * blockDim.x + threadIdx.x;
    if (i < n) dst[i] = a[i] + b[i];
}

// ---------------------------------------------------------------------------
// Persistent LSTM kernel.
// WG w owns hidden tile n16=w for layers 1,2 (32 tiles each) and, for w<16,
// tile n16=w for layer 3. Waves 0-3: layer1 (+layer3), m = wave; waves 4-7:
// layer2, m = wave-4. Cell state c lives in registers for the whole sequence.
// ---------------------------------------------------------------------------
__global__ __launch_bounds__(WGSIZE) void lstm_persistent(
    const float* __restrict__ seq, const unsigned char* __restrict__ mask,
    const half_t* __restrict__ wB1, const half_t* __restrict__ wB2,
    const half_t* __restrict__ wB3,
    const float* __restrict__ bias1, const float* __restrict__ bias2,
    const float* __restrict__ bias3,
    half_t* __restrict__ h1buf, half_t* __restrict__ h2buf,
    half_t* __restrict__ h3buf,
    unsigned* __restrict__ bar, float* __restrict__ out) {
    extern __shared__ char smem_raw[];
    half_t* lds = (half_t*)smem_raw;

    const int tid  = threadIdx.x;
    const int lane = tid & 31;
    const int wave = tid >> 5;
    const int wg   = blockIdx.x;

    half_t* ldsB1 = lds;                                  // [4][KS1][512]  96KB
    half_t* ldsB2 = lds + 4 * KS1 * 512;                  // [4][KS2][512] 128KB
    half_t* ldsB3 = lds + 4 * KS1 * 512 + 4 * KS2 * 512;  // [4][KS3][512]  96KB

    // ---- one-time: pin this WG's weight shard in LDS ----
#pragma unroll
    for (int g = 0; g < 4; ++g) {
        const v8h* src = (const v8h*)(wB1 + (size_t)(g * 32 + wg) * KS1 * 512);
        v8h* dst = (v8h*)(ldsB1 + g * KS1 * 512);
        for (int i = tid; i < KS1 * 512 / 8; i += WGSIZE) dst[i] = src[i];
    }
#pragma unroll
    for (int g = 0; g < 4; ++g) {
        const v8h* src = (const v8h*)(wB2 + (size_t)(g * 32 + wg) * KS2 * 512);
        v8h* dst = (v8h*)(ldsB2 + g * KS2 * 512);
        for (int i = tid; i < KS2 * 512 / 8; i += WGSIZE) dst[i] = src[i];
    }
    if (wg < 16) {
#pragma unroll
        for (int g = 0; g < 4; ++g) {
            const v8h* src = (const v8h*)(wB3 + (size_t)(g * 16 + wg) * KS3 * 512);
            v8h* dst = (v8h*)(ldsB3 + g * KS3 * 512);
            for (int i = tid; i < KS3 * 512 / 8; i += WGSIZE) dst[i] = src[i];
        }
    }
    __syncthreads();

    const int  nloc  = lane & 15;              // column within 16-wide tile
    const int  rofs  = (lane & 16) ? 8 : 0;    // C/D row offset for this lane
    const bool isL1  = (wave < 4);
    const bool isL3  = isL1 && (wg < 16);
    const int  mrow1 = (wave & 3) * 16;        // batch-row base, layer1/3 waves
    const int  mrow2 = ((wave - 4) & 3) * 16;  // batch-row base, layer2 waves
    const int  col   = wg * 16 + nloc;         // global hidden column

    float bi1[4], bi2[4], bi3[4];
#pragma unroll
    for (int g = 0; g < 4; ++g) {
        bi1[g] = bias1[g * NHID + col];
        bi2[g] = bias2[g * NHID + col];
        bi3[g] = (wg < 16) ? bias3[g * NOUT + col] : 0.0f;
    }
    float c1[8], c2[8], c3[8];
#pragma unroll
    for (int r = 0; r < 8; ++r) { c1[r] = 0.f; c2[r] = 0.f; c3[r] = 0.f; }

    unsigned bar_target = 0;

    for (int t = 0; t < T_STEPS; ++t) {
        const int p = t & 1;  // read index; write index = 1-p (ping-pong)
        half_t* h1p = h1buf + (size_t)p * BATCH * NHID;
        half_t* h1c = h1buf + (size_t)(1 - p) * BATCH * NHID;
        half_t* h2p = h2buf + (size_t)p * BATCH * NHID;
        half_t* h2c = h2buf + (size_t)(1 - p) * BATCH * NHID;
        half_t* h3p = h3buf + (size_t)p * BATCH * NOUT;
        half_t* h3c = h3buf + (size_t)(1 - p) * BATCH * NOUT;
        const bool use_seq = (t == 0) || (mask[t] != 0);
        const bool last = (t == T_STEPS - 1);

        // ---------------- layer 1: x(+teacher-force) | h1_prev ----------------
        if (isL1) {
            const int arow = mrow1 + nloc;
            v8f acc[4] = {acc_zero(), acc_zero(), acc_zero(), acc_zero()};
            // x part (K 0..255): branch hoisted out of the K-loop (wave-uniform)
            if (use_seq) {
                const float* xs = seq + (size_t)t * BATCH * NIN;
                for (int ks = 0; ks < NIN / 32; ++ks) {
                    v16h a = load_a_f32(xs, NIN, arow, ks * 32, lane);
#pragma unroll
                    for (int g = 0; g < 4; ++g)
                        acc[g] = wmma_step(a, load_b_lds(ldsB1 + (g * KS1 + ks) * 512, lane), acc[g]);
                }
            } else {
                for (int ks = 0; ks < NIN / 32; ++ks) {
                    v16h a = load_a_f16(h3p, NOUT, arow, ks * 32, lane);
#pragma unroll
                    for (int g = 0; g < 4; ++g)
                        acc[g] = wmma_step(a, load_b_lds(ldsB1 + (g * KS1 + ks) * 512, lane), acc[g]);
                }
            }
            // h1_prev part (K 256..767)
            for (int ks = 0; ks < NHID / 32; ++ks) {
                v16h a = load_a_f16(h1p, NHID, arow, ks * 32, lane);
#pragma unroll
                for (int g = 0; g < 4; ++g)
                    acc[g] = wmma_step(a, load_b_lds(ldsB1 + (g * KS1 + 8 + ks) * 512, lane), acc[g]);
            }
#pragma unroll
            for (int r = 0; r < 8; ++r) {
                float iv = sigf(acc[0][r] + bi1[0]);
                float fv = sigf(acc[1][r] + bi1[1]);
                float gv = tanh_fast(acc[2][r] + bi1[2]);
                float ov = sigf(acc[3][r] + bi1[3]);
                c1[r] = fv * c1[r] + iv * gv;
                float h = ov * tanh_fast(c1[r]);
                int row = mrow1 + r + rofs;
                h1c[(size_t)row * NHID + col] = (half_t)h;
                if (last) out[OFF_H1 + (size_t)row * NHID + col] = h;
            }
        }
        grid_sync(bar, &bar_target);

        // ---------------- layer 2: h1_new | h2_prev ----------------
        if (!isL1) {
            const int arow = mrow2 + nloc;
            v8f acc[4] = {acc_zero(), acc_zero(), acc_zero(), acc_zero()};
            for (int ks = 0; ks < NHID / 32; ++ks) {
                v16h a = load_a_f16(h1c, NHID, arow, ks * 32, lane);
#pragma unroll
                for (int g = 0; g < 4; ++g)
                    acc[g] = wmma_step(a, load_b_lds(ldsB2 + (g * KS2 + ks) * 512, lane), acc[g]);
            }
            for (int ks = 0; ks < NHID / 32; ++ks) {
                v16h a = load_a_f16(h2p, NHID, arow, ks * 32, lane);
#pragma unroll
                for (int g = 0; g < 4; ++g)
                    acc[g] = wmma_step(a, load_b_lds(ldsB2 + (g * KS2 + 16 + ks) * 512, lane), acc[g]);
            }
#pragma unroll
            for (int r = 0; r < 8; ++r) {
                float iv = sigf(acc[0][r] + bi2[0]);
                float fv = sigf(acc[1][r] + bi2[1]);
                float gv = tanh_fast(acc[2][r] + bi2[2]);
                float ov = sigf(acc[3][r] + bi2[3]);
                c2[r] = fv * c2[r] + iv * gv;
                float h = ov * tanh_fast(c2[r]);
                int row = mrow2 + r + rofs;
                h2c[(size_t)row * NHID + col] = (half_t)h;
                if (last) out[OFF_H2 + (size_t)row * NHID + col] = h;
            }
        }
        grid_sync(bar, &bar_target);

        // ---------------- layer 3: h2_new | h3_prev ----------------
        if (isL3) {
            const int arow = mrow1 + nloc;
            v8f acc[4] = {acc_zero(), acc_zero(), acc_zero(), acc_zero()};
            for (int ks = 0; ks < NHID / 32; ++ks) {
                v16h a = load_a_f16(h2c, NHID, arow, ks * 32, lane);
#pragma unroll
                for (int g = 0; g < 4; ++g)
                    acc[g] = wmma_step(a, load_b_lds(ldsB3 + (g * KS3 + ks) * 512, lane), acc[g]);
            }
            for (int ks = 0; ks < NOUT / 32; ++ks) {
                v16h a = load_a_f16(h3p, NOUT, arow, ks * 32, lane);
#pragma unroll
                for (int g = 0; g < 4; ++g)
                    acc[g] = wmma_step(a, load_b_lds(ldsB3 + (g * KS3 + 16 + ks) * 512, lane), acc[g]);
            }
#pragma unroll
            for (int r = 0; r < 8; ++r) {
                float iv = sigf(acc[0][r] + bi3[0]);
                float fv = sigf(acc[1][r] + bi3[1]);
                float gv = tanh_fast(acc[2][r] + bi3[2]);
                float ov = sigf(acc[3][r] + bi3[3]);
                c3[r] = fv * c3[r] + iv * gv;
                float h = ov * tanh_fast(c3[r]);
                int row = mrow1 + r + rofs;
                h3c[(size_t)row * NOUT + col] = (half_t)h;
                out[(size_t)t * BATCH * NOUT + (size_t)row * NOUT + col] = h;
                if (last) out[OFF_H3 + (size_t)row * NOUT + col] = h;
            }
        }
        grid_sync(bar, &bar_target);
    }

    // ---- final cell states ----
    if (isL1)
#pragma unroll
        for (int r = 0; r < 8; ++r)
            out[OFF_C1 + (size_t)(mrow1 + r + rofs) * NHID + col] = c1[r];
    if (!isL1)
#pragma unroll
        for (int r = 0; r < 8; ++r)
            out[OFF_C2 + (size_t)(mrow2 + r + rofs) * NHID + col] = c2[r];
    if (isL3)
#pragma unroll
        for (int r = 0; r < 8; ++r)
            out[OFF_C3 + (size_t)(mrow1 + r + rofs) * NOUT + col] = c3[r];
}

// ---------------------------------------------------------------------------
extern "C" void kernel_launch(void* const* d_in, const int* in_sizes, int n_in,
                              void* d_out, int out_size, void* d_ws, size_t ws_size,
                              hipStream_t stream) {
    const float*         seq  = (const float*)d_in[0];
    const unsigned char* mask = (const unsigned char*)d_in[1];
    const float* Wih1 = (const float*)d_in[2];
    const float* Whh1 = (const float*)d_in[3];
    const float* bih1 = (const float*)d_in[4];
    const float* bhh1 = (const float*)d_in[5];
    const float* Wih2 = (const float*)d_in[6];
    const float* Whh2 = (const float*)d_in[7];
    const float* bih2 = (const float*)d_in[8];
    const float* bhh2 = (const float*)d_in[9];
    const float* Wih3 = (const float*)d_in[10];
    const float* Whh3 = (const float*)d_in[11];
    const float* bih3 = (const float*)d_in[12];
    const float* bhh3 = (const float*)d_in[13];
    float* out = (float*)d_out;

    // workspace carve-up (256B aligned)
    char* ws = (char*)d_ws;
    size_t off = 0;
    auto carve = [&](size_t bytes) -> void* {
        void* p = ws + off;
        off = (off + bytes + 255) & ~(size_t)255;
        return p;
    };
    const size_t nB1 = (size_t)128 * KS1 * 512;  // halves
    const size_t nB2 = (size_t)128 * KS2 * 512;
    const size_t nB3 = (size_t)64 * KS3 * 512;
    half_t* wB1   = (half_t*)carve(nB1 * sizeof(half_t));
    half_t* wB2   = (half_t*)carve(nB2 * sizeof(half_t));
    half_t* wB3   = (half_t*)carve(nB3 * sizeof(half_t));
    float*  bias1 = (float*)carve(4 * NHID * sizeof(float));
    float*  bias2 = (float*)carve(4 * NHID * sizeof(float));
    float*  bias3 = (float*)carve(4 * NOUT * sizeof(float));
    // contiguous zero-init region: h ping-pong buffers + barrier counter
    char*   zbase = ws + off;
    half_t* h1buf = (half_t*)carve(2 * BATCH * NHID * sizeof(half_t));
    half_t* h2buf = (half_t*)carve(2 * BATCH * NHID * sizeof(half_t));
    half_t* h3buf = (half_t*)carve(2 * BATCH * NOUT * sizeof(half_t));
    unsigned* bar = (unsigned*)carve(256);
    size_t zbytes = (size_t)((char*)ws + off - zbase);

    hipMemsetAsync(zbase, 0, zbytes, stream);

    // repack weights into LDS-swizzled WMMA-B layout (f16) + fold biases
    repack_weights<<<(unsigned)((nB1 + 255) / 256), 256, 0, stream>>>(
        Wih1, Whh1, NIN, NHID, 128, wB1);
    repack_weights<<<(unsigned)((nB2 + 255) / 256), 256, 0, stream>>>(
        Wih2, Whh2, NHID, NHID, 128, wB2);
    repack_weights<<<(unsigned)((nB3 + 255) / 256), 256, 0, stream>>>(
        Wih3, Whh3, NHID, NOUT, 64, wB3);
    bias_sum<<<(4 * NHID + 255) / 256, 256, 0, stream>>>(bih1, bhh1, bias1, 4 * NHID);
    bias_sum<<<(4 * NHID + 255) / 256, 256, 0, stream>>>(bih2, bhh2, bias2, 4 * NHID);
    bias_sum<<<(4 * NOUT + 255) / 256, 256, 0, stream>>>(bih3, bhh3, bias3, 4 * NOUT);

    // persistent sequence kernel: 32 WGs, 8 waves each, 320KB LDS per WG
    lstm_persistent<<<NWG, WGSIZE, 320 * 1024, stream>>>(
        seq, mask, wB1, wB2, wB3, bias1, bias2, bias3,
        h1buf, h2buf, h3buf, bar, out);
}
